// GCN_LSTM_Attn_240518168639
// MI455X (gfx1250) — matrix operand order repaired and checked by reading, not compiled, not run on hardware
//
#include <hip/hip_runtime.h>

// ---------------------------------------------------------------------------
// GCN -> LSTM -> Attention pipeline for MI455X (gfx1250, wave32, WMMA bf16)
// B=8, T=12, N=2000, F=16, H=64, E=16000
// ---------------------------------------------------------------------------

#define Bq   8
#define Tq   12
#define Nq   2000
#define Fq   16
#define Hq   64
#define Eq   16000
#define MGq  (Bq*Tq)       /* 96 graphs            */
#define ROWSq (MGq*Nq)     /* 192000 GCN rows      */
#define LROWSq (Bq*Nq)     /* 16000 LSTM rows      */

typedef __attribute__((ext_vector_type(16))) __bf16 v16bf;   // 32 B (one WMMA A/B operand)
typedef __attribute__((ext_vector_type(8)))  __bf16 v8bf;    // 16 B
typedef __attribute__((ext_vector_type(8)))  float  v8f;     // WMMA C/D operand

__device__ __forceinline__ float sigf(float x) { return 1.0f / (1.0f + __expf(-x)); }

// ------------------------- normalization prep ------------------------------
__global__ void k_deg_init(float* deg) {
    int i = blockIdx.x * blockDim.x + threadIdx.x;
    if (i < Nq) deg[i] = 1.0f;
}

__global__ void k_deg_acc(const long long* __restrict__ ei, float* deg) {
    int e = blockIdx.x * blockDim.x + threadIdx.x;
    if (e < Eq) atomicAdd(&deg[(int)ei[Eq + e]], 1.0f);
}

__global__ void k_norms(const float* __restrict__ deg, float* dinv, float* nself) {
    int i = blockIdx.x * blockDim.x + threadIdx.x;
    if (i < Nq) {
        float d = deg[i];
        dinv[i]  = rsqrtf(d);
        nself[i] = 1.0f / d;
    }
}

__global__ void k_norme(const long long* __restrict__ ei, const float* __restrict__ dinv,
                        float* ne) {
    int e = blockIdx.x * blockDim.x + threadIdx.x;
    if (e < Eq) ne[e] = dinv[(int)ei[e]] * dinv[(int)ei[Eq + e]];
}

// ---- pack a [KIN x 64] f32 weight into WMMA-B lane layout (bf16) ----------
// P[((kt*4 + nt)*32 + lane)*16 + j] = W[(kt*32 + (j>>3)*16 + half*8 + (j&7)) * 64 + nt*16 + nn]
__global__ void k_pack_w(const float* __restrict__ W, __bf16* __restrict__ P, int KIN) {
    int i = blockIdx.x * blockDim.x + threadIdx.x;
    int total = ((KIN + 31) / 32) * 4 * 32 * 16;
    if (i >= total) return;
    int j = i & 15, lane = (i >> 4) & 31, ntkt = i >> 9;
    int nt = ntkt & 3, kt = ntkt >> 2;
    int half = lane >> 4, nn = lane & 15;
    int k = kt * 32 + (j >> 3) * 16 + half * 8 + (j & 7), col = nt * 16 + nn;
    P[i] = (k < KIN) ? (__bf16)W[k * Hq + col] : (__bf16)0.0f;
}

// ---- pack LSTM weights [W_ih; W_hh]^T (K=128, Ncols=256) into B layout ----
__global__ void k_pack_lstm(const float* __restrict__ Wih, const float* __restrict__ Whh,
                            const float* __restrict__ bih, const float* __restrict__ bhh,
                            __bf16* __restrict__ P, float* __restrict__ bc) {
    int i = blockIdx.x * blockDim.x + threadIdx.x;
    if (i < 256) bc[i] = bih[i] + bhh[i];
    if (i >= 4 * 16 * 32 * 16) return;
    int j = i & 15, lane = (i >> 4) & 31, ntkt = i >> 9;
    int nt = ntkt & 15, kt = ntkt >> 4;
    int half = lane >> 4, nn = lane & 15;
    int k = kt * 32 + (j >> 3) * 16 + half * 8 + (j & 7);     // 0..127 over [x | h]
    int col = nt * 16 + nn;              // 0..255 gate column
    float w = (k < Hq) ? Wih[col * Hq + k] : Whh[col * Hq + (k - Hq)];
    P[i] = (__bf16)w;
}

// ------------------------- GCN dense linear (WMMA) --------------------------
// Y[r,0..63] = X[r,0..KIN-1] @ W[KIN x 64].  One wave = one 16-row tile.
// KIN is a template constant -> zero-padding folds at compile time.
template <int KIN>
__global__ __launch_bounds__(256) void k_gcn_mm(const float* __restrict__ X,
                                                const __bf16* __restrict__ Wp,
                                                float* __restrict__ Y, int nrows) {
    constexpr int NKT = (KIN + 31) / 32;
    int wave = (int)((blockIdx.x * (unsigned)blockDim.x + threadIdx.x) >> 5);
    int lane = threadIdx.x & 31;
    int r0 = wave * 16;
    if (r0 >= nrows) return;
    int half = lane >> 4, nn = lane & 15;

    const v16bf* Bp = (const v16bf*)Wp;

    v8f acc[4];
#pragma unroll
    for (int nt = 0; nt < 4; ++nt) acc[nt] = (v8f){0, 0, 0, 0, 0, 0, 0, 0};

#pragma unroll
    for (int kt = 0; kt < NKT; ++kt) {
        // A operand: row m = r0+nn; element j -> k = kt*32 + (j<8?0:16) + 8*half + (j&7)
        v16bf a;
        const float* xr = X + (size_t)(r0 + nn) * KIN;
#pragma unroll
        for (int g = 0; g < 2; ++g) {
            if (kt * 32 + g * 16 + 15 < KIN) {   // compile-time (uniform in half)
                int kb = kt * 32 + g * 16 + half * 8;
                float4 u = *reinterpret_cast<const float4*>(xr + kb);
                float4 v = *reinterpret_cast<const float4*>(xr + kb + 4);
                a[g * 8 + 0] = (__bf16)u.x; a[g * 8 + 1] = (__bf16)u.y;
                a[g * 8 + 2] = (__bf16)u.z; a[g * 8 + 3] = (__bf16)u.w;
                a[g * 8 + 4] = (__bf16)v.x; a[g * 8 + 5] = (__bf16)v.y;
                a[g * 8 + 6] = (__bf16)v.z; a[g * 8 + 7] = (__bf16)v.w;
            } else {
#pragma unroll
                for (int j = 0; j < 8; ++j) a[g * 8 + j] = (__bf16)0.0f;
            }
        }
#pragma unroll
        for (int nt = 0; nt < 4; ++nt) {
            v16bf b = Bp[(kt * 4 + nt) * 32 + lane];   // pre-packed B operand
            acc[nt] = __builtin_amdgcn_wmma_f32_16x16x32_bf16(
                false, a, false, b, (short)0, acc[nt], false, false);
        }
    }
#pragma unroll
    for (int nt = 0; nt < 4; ++nt)
#pragma unroll
        for (int i = 0; i < 8; ++i)
            Y[(size_t)(r0 + 8 * half + i) * Hq + nt * 16 + nn] = acc[nt][i];
}

// ------------------------- GCN aggregation ---------------------------------
__global__ void k_agg_init(const float* __restrict__ lin, const float* __restrict__ nself,
                           const float* __restrict__ bias, float* __restrict__ out) {
    long long i = (long long)blockIdx.x * blockDim.x + threadIdx.x;
    if (i >= (long long)ROWSq * 16) return;
    int c4 = (int)(i & 15);
    long long rn = i >> 4;              // g*Nq + n
    int n = (int)(rn % Nq);
    const float4 v = *reinterpret_cast<const float4*>(lin + rn * Hq + c4 * 4);
    float s = nself[n];
    float4 o;
    o.x = v.x * s + bias[c4 * 4 + 0];
    o.y = v.y * s + bias[c4 * 4 + 1];
    o.z = v.z * s + bias[c4 * 4 + 2];
    o.w = v.w * s + bias[c4 * 4 + 3];
    *reinterpret_cast<float4*>(out + rn * Hq + c4 * 4) = o;
}

__global__ void k_agg_edge(const float* __restrict__ lin, const long long* __restrict__ ei,
                           const float* __restrict__ ne, float* __restrict__ out) {
    long long t = (long long)blockIdx.x * blockDim.x + threadIdx.x;
    if (t >= (long long)MGq * Eq * 16) return;
    int c4 = (int)(t & 15);
    long long ge = t >> 4;
    int e = (int)(ge % Eq);
    int g = (int)(ge / Eq);
    int s = (int)ei[e], d = (int)ei[Eq + e];
    float w = ne[e];
    const float* p = lin + ((size_t)g * Nq + s) * Hq + c4 * 4;
    float* q = out + ((size_t)g * Nq + d) * Hq + c4 * 4;
    atomicAdd(q + 0, p[0] * w);
    atomicAdd(q + 1, p[1] * w);
    atomicAdd(q + 2, p[2] * w);
    atomicAdd(q + 3, p[3] * w);
}

__global__ void k_relu(float* __restrict__ buf) {
    long long i = (long long)blockIdx.x * blockDim.x + threadIdx.x;
    if (i >= (long long)ROWSq * 16) return;
    float4 v = *reinterpret_cast<float4*>(buf + i * 4);
    v.x = fmaxf(v.x, 0.0f); v.y = fmaxf(v.y, 0.0f);
    v.z = fmaxf(v.z, 0.0f); v.w = fmaxf(v.w, 0.0f);
    *reinterpret_cast<float4*>(buf + i * 4) = v;
}

// layer-2 aggregation writes straight into LSTM layout [B*N][T][H] (fused transpose)
__global__ void k_agg_init2(const float* __restrict__ lin, const float* __restrict__ nself,
                            const float* __restrict__ bias, float* __restrict__ out) {
    long long i = (long long)blockIdx.x * blockDim.x + threadIdx.x;
    if (i >= (long long)ROWSq * 16) return;
    int c4 = (int)(i & 15);
    long long rn = i >> 4;             // g*Nq + n
    int n = (int)(rn % Nq);
    int g = (int)(rn / Nq);
    int b = g / Tq, t = g % Tq;
    const float4 v = *reinterpret_cast<const float4*>(lin + rn * Hq + c4 * 4);
    float s = nself[n];
    float4 o;
    o.x = v.x * s + bias[c4 * 4 + 0];
    o.y = v.y * s + bias[c4 * 4 + 1];
    o.z = v.z * s + bias[c4 * 4 + 2];
    o.w = v.w * s + bias[c4 * 4 + 3];
    size_t dst = ((size_t)((size_t)b * Nq + n) * Tq + t) * Hq + c4 * 4;
    *reinterpret_cast<float4*>(out + dst) = o;
}

__global__ void k_agg_edge2(const float* __restrict__ lin, const long long* __restrict__ ei,
                            const float* __restrict__ ne, float* __restrict__ out) {
    long long tt = (long long)blockIdx.x * blockDim.x + threadIdx.x;
    if (tt >= (long long)MGq * Eq * 16) return;
    int c4 = (int)(tt & 15);
    long long ge = tt >> 4;
    int e = (int)(ge % Eq);
    int g = (int)(ge / Eq);
    int b = g / Tq, t = g % Tq;
    int s = (int)ei[e], d = (int)ei[Eq + e];
    float w = ne[e];
    const float* p = lin + ((size_t)g * Nq + s) * Hq + c4 * 4;
    float* q = out + ((size_t)((size_t)b * Nq + d) * Tq + t) * Hq + c4 * 4;
    atomicAdd(q + 0, p[0] * w);
    atomicAdd(q + 1, p[1] * w);
    atomicAdd(q + 2, p[2] * w);
    atomicAdd(q + 3, p[3] * w);
}

// ------------------------- fused LSTM + attention ---------------------------
// One 256-thread block (8 waves) owns 16 rows for all 12 timesteps.
__global__ __launch_bounds__(256) void k_lstm_attn(
    const float* __restrict__ Xin,       // [LROWS][T][H]
    const __bf16* __restrict__ Wp,       // packed B layout [4 kt][16 nt][32 lane][16]
    const float* __restrict__ bc,        // [256]
    const float* __restrict__ attn_w, const float* __restrict__ attn_b,
    const float* __restrict__ fc_w, const float* __restrict__ fc_b,
    float* __restrict__ out)             // [LROWS]
{
    __shared__ float  g_lds[16][256];                       // 16 KB
    __shared__ alignas(16) __bf16 hall[Tq][16][Hq];         // 24 KB

    int tid = threadIdx.x;
    int wave = tid >> 5, lane = tid & 31;
    int half = lane >> 4, nn = lane & 15;
    int r0 = blockIdx.x * 16;

    const v16bf* Bp = (const v16bf*)Wp;

    // weights: wave w owns N-tiles 2w, 2w+1; 4 K-tiles of 32 each (one vec load each)
    v16bf bw[2][4];
#pragma unroll
    for (int p = 0; p < 2; ++p) {
        int nt = 2 * wave + p;
#pragma unroll
        for (int kt = 0; kt < 4; ++kt)
            bw[p][kt] = Bp[(kt * 16 + nt) * 32 + lane];
    }

    float cstate[4] = {0.f, 0.f, 0.f, 0.f};

    for (int t = 0; t < Tq; ++t) {
        // A from x_t (K 0..63), vectorized float4 loads
        v16bf ax[2];
        {
            const float* xr = Xin + ((size_t)(r0 + nn) * Tq + t) * Hq;
#pragma unroll
            for (int kt = 0; kt < 2; ++kt)
#pragma unroll
                for (int g = 0; g < 2; ++g) {
                    int kb = kt * 32 + g * 16 + half * 8;
                    float4 u = *reinterpret_cast<const float4*>(xr + kb);
                    float4 v = *reinterpret_cast<const float4*>(xr + kb + 4);
                    ax[kt][g * 8 + 0] = (__bf16)u.x; ax[kt][g * 8 + 1] = (__bf16)u.y;
                    ax[kt][g * 8 + 2] = (__bf16)u.z; ax[kt][g * 8 + 3] = (__bf16)u.w;
                    ax[kt][g * 8 + 4] = (__bf16)v.x; ax[kt][g * 8 + 5] = (__bf16)v.y;
                    ax[kt][g * 8 + 6] = (__bf16)v.z; ax[kt][g * 8 + 7] = (__bf16)v.w;
                }
        }
        v8f acc0 = (v8f){0, 0, 0, 0, 0, 0, 0, 0};
        v8f acc1 = (v8f){0, 0, 0, 0, 0, 0, 0, 0};
        acc0 = __builtin_amdgcn_wmma_f32_16x16x32_bf16(false, ax[0], false, bw[0][0], (short)0, acc0, false, false);
        acc0 = __builtin_amdgcn_wmma_f32_16x16x32_bf16(false, ax[1], false, bw[0][1], (short)0, acc0, false, false);
        acc1 = __builtin_amdgcn_wmma_f32_16x16x32_bf16(false, ax[0], false, bw[1][0], (short)0, acc1, false, false);
        acc1 = __builtin_amdgcn_wmma_f32_16x16x32_bf16(false, ax[1], false, bw[1][1], (short)0, acc1, false, false);
        if (t > 0) {   // A from h_{t-1} (K 64..127), 16-byte LDS vector loads
            v16bf ah[2];
            const v8bf* hp = reinterpret_cast<const v8bf*>(&hall[t - 1][nn][0]);
#pragma unroll
            for (int kt = 0; kt < 2; ++kt)
#pragma unroll
                for (int g = 0; g < 2; ++g) {
                    int kb = kt * 32 + g * 16 + half * 8;
                    v8bf h8 = hp[kb >> 3];
#pragma unroll
                    for (int j = 0; j < 8; ++j) ah[kt][g * 8 + j] = h8[j];
                }
            acc0 = __builtin_amdgcn_wmma_f32_16x16x32_bf16(false, ah[0], false, bw[0][2], (short)0, acc0, false, false);
            acc0 = __builtin_amdgcn_wmma_f32_16x16x32_bf16(false, ah[1], false, bw[0][3], (short)0, acc0, false, false);
            acc1 = __builtin_amdgcn_wmma_f32_16x16x32_bf16(false, ah[0], false, bw[1][2], (short)0, acc1, false, false);
            acc1 = __builtin_amdgcn_wmma_f32_16x16x32_bf16(false, ah[1], false, bw[1][3], (short)0, acc1, false, false);
        }
        {
            int nt0 = 2 * wave, nt1 = 2 * wave + 1;
#pragma unroll
            for (int i = 0; i < 8; ++i) {
                g_lds[8 * half + i][nt0 * 16 + nn] = acc0[i];
                g_lds[8 * half + i][nt1 * 16 + nn] = acc1[i];
            }
        }
        __syncthreads();
        // gate nonlinearity; each thread owns 4 (m,h) cells persistently
#pragma unroll
        for (int q = 0; q < 4; ++q) {
            int idx = tid + q * 256;
            int m = idx >> 6, hh = idx & 63;
            float gi = g_lds[m][hh]        + bc[hh];
            float gf = g_lds[m][64 + hh]   + bc[64 + hh];
            float gg = g_lds[m][128 + hh]  + bc[128 + hh];
            float go = g_lds[m][192 + hh]  + bc[192 + hh];
            float c  = sigf(gf) * cstate[q] + sigf(gi) * tanhf(gg);
            cstate[q] = c;
            hall[t][m][hh] = (__bf16)(sigf(go) * tanhf(c));
        }
        __syncthreads();
    }

    // attention pooling over time + final FC, 1 thread per row
    if (tid < 16) {
        int m = tid;
        float sc[Tq], mx = -1e30f;
        for (int t = 0; t < Tq; ++t) {
            float s = attn_b[0];
            for (int hh = 0; hh < Hq; ++hh) s += (float)hall[t][m][hh] * attn_w[hh];
            sc[t] = s;
            mx = fmaxf(mx, s);
        }
        float denom = 0.f;
        for (int t = 0; t < Tq; ++t) { sc[t] = __expf(sc[t] - mx); denom += sc[t]; }
        float inv = 1.0f / denom;
        float o = fc_b[0];
        for (int hh = 0; hh < Hq; ++hh) {
            float ctx = 0.f;
            for (int t = 0; t < Tq; ++t) ctx += sc[t] * (float)hall[t][m][hh];
            o += ctx * inv * fc_w[hh];
        }
        out[r0 + m] = o;
    }
}

// ---------------------------------------------------------------------------
extern "C" void kernel_launch(void* const* d_in, const int* in_sizes, int n_in,
                              void* d_out, int out_size, void* d_ws, size_t ws_size,
                              hipStream_t stream) {
    const float* x      = (const float*)d_in[0];
    const float* W1     = (const float*)d_in[1];
    const float* b1     = (const float*)d_in[2];
    const float* W2     = (const float*)d_in[3];
    const float* b2     = (const float*)d_in[4];
    const float* W_ih   = (const float*)d_in[5];
    const float* W_hh   = (const float*)d_in[6];
    const float* b_ih   = (const float*)d_in[7];
    const float* b_hh   = (const float*)d_in[8];
    const float* attn_w = (const float*)d_in[9];
    const float* attn_b = (const float*)d_in[10];
    const float* fc_w   = (const float*)d_in[11];
    const float* fc_b   = (const float*)d_in[12];
    const long long* ei = (const long long*)d_in[13];
    float* out = (float*)d_out;

    char* ws = (char*)d_ws;
    float*  deg   = (float*)(ws + 0);
    float*  dinv  = (float*)(ws + 8192);
    float*  nself = (float*)(ws + 16384);
    float*  ne    = (float*)(ws + 24576);          // 64 KB
    float*  bc    = (float*)(ws + 90112);          // 1 KB
    __bf16* WpL   = (__bf16*)(ws + 91136);         // LSTM packed: 32768 bf16 = 64 KB
    __bf16* Wp1   = (__bf16*)(ws + 156672);        // 2048 bf16 = 4 KB
    __bf16* Wp2   = (__bf16*)(ws + 160768);        // 4096 bf16 = 8 KB
    float*  bufA  = (float*)(ws + 168960);                          // 49.152 MB
    float*  bufB  = (float*)(ws + 168960 + (size_t)ROWSq * Hq * 4); // 49.152 MB

    // normalization + weight packing
    k_deg_init<<<(Nq + 255) / 256, 256, 0, stream>>>(deg);
    k_deg_acc<<<(Eq + 255) / 256, 256, 0, stream>>>(ei, deg);
    k_norms<<<(Nq + 255) / 256, 256, 0, stream>>>(deg, dinv, nself);
    k_norme<<<(Eq + 255) / 256, 256, 0, stream>>>(ei, dinv, ne);
    k_pack_w<<<(2048 + 255) / 256, 256, 0, stream>>>(W1, Wp1, Fq);
    k_pack_w<<<(4096 + 255) / 256, 256, 0, stream>>>(W2, Wp2, Hq);
    k_pack_lstm<<<(32768) / 256, 256, 0, stream>>>(W_ih, W_hh, b_ih, b_hh, WpL, bc);

    const long long n4  = (long long)ROWSq * 16;        // float4 units
    const long long ne4 = (long long)MGq * Eq * 16;     // edge float4 work items

    // GCN layer 1
    k_gcn_mm<Fq><<<ROWSq / 16 / 8, 256, 0, stream>>>(x, Wp1, bufA, ROWSq);
    k_agg_init<<<(unsigned)((n4 + 255) / 256), 256, 0, stream>>>(bufA, nself, b1, bufB);
    k_agg_edge<<<(unsigned)((ne4 + 255) / 256), 256, 0, stream>>>(bufA, ei, ne, bufB);
    k_relu<<<(unsigned)((n4 + 255) / 256), 256, 0, stream>>>(bufB);

    // GCN layer 2 (aggregation writes LSTM layout directly)
    k_gcn_mm<Hq><<<ROWSq / 16 / 8, 256, 0, stream>>>(bufB, Wp2, bufA, ROWSq);
    k_agg_init2<<<(unsigned)((n4 + 255) / 256), 256, 0, stream>>>(bufA, nself, b2, bufB);
    k_agg_edge2<<<(unsigned)((ne4 + 255) / 256), 256, 0, stream>>>(bufA, ei, ne, bufB);

    // fused LSTM + attention + FC
    k_lstm_attn<<<LROWSq / 16, 256, 0, stream>>>(bufB, WpL, bc, attn_w, attn_b,
                                                 fc_w, fc_b, out);
}
